// STEGMN_28432683499985
// MI455X (gfx1250) — compile-verified, matrix operand order checked by reading, not run on hardware
//
#include <hip/hip_runtime.h>
#include <hip/hip_bf16.h>
#include <math.h>

// Problem constants (from reference)
#define T_ 10
#define N_ 5000
#define E_ 80000
#define H_ 128
#define IN_ 16
#define F_ 1

typedef __attribute__((ext_vector_type(16))) __bf16 v16bf;
typedef __attribute__((ext_vector_type(8)))  float  v8f;

__device__ __forceinline__ float silu_f(float z){ return z / (1.0f + __expf(-z)); }

// K-pair base for WMMA 16-bit A/B fragment element pairs (CDNA5 layout):
// lane<16 holds K {0..7, 16..23}; lane>=16 holds K {8..15, 24..31} (per 32-K block kb)
__device__ __forceinline__ int frag_k0(int kb, int half, int j){
  return kb*32 + ((j < 4) ? (half*8 + 2*j) : (16 + half*8 + 2*(j-4)));
}

__device__ __forceinline__ v16bf afrag_lds_bf16(const __bf16* s, int lane, int kb){
  int m = lane & 15, half = lane >> 4;
  v16bf a;
#pragma unroll
  for (int j = 0; j < 8; ++j){
    int k0 = frag_k0(kb, half, j);
    a[2*j]   = s[m*H_ + k0];
    a[2*j+1] = s[m*H_ + k0 + 1];
  }
  return a;
}

__device__ __forceinline__ v16bf afrag_lds_f32(const float* s, int lane, int kb){
  int m = lane & 15, half = lane >> 4;
  v16bf a;
#pragma unroll
  for (int j = 0; j < 8; ++j){
    int k0 = frag_k0(kb, half, j);
    a[2*j]   = (__bf16)s[m*H_ + k0];
    a[2*j+1] = (__bf16)s[m*H_ + k0 + 1];
  }
  return a;
}

// B fragments are pre-packed contiguously: [(kb*8+jn)*32 + lane] * 16 bf16 elements
__device__ __forceinline__ v16bf bfrag(const unsigned short* wp, int lane, int kb, int jn){
  const v16bf* p = (const v16bf*)wp;
  return p[(size_t)((kb*8 + jn)*32 + lane)];
}

__device__ __forceinline__ v8f wmma_acc(v16bf a, v16bf b, v8f c){
  return __builtin_amdgcn_wmma_f32_16x16x32_bf16(false, a, false, b, (short)0, c, false, false);
}

// ---------------------------------------------------------------------------
// Pack a fp32 weight [K x 128] (row-major, K = KB*32) into WMMA B-fragment
// order in bf16: dst[((kb*8+jn)*32+lane)*16 + i]
// ---------------------------------------------------------------------------
__global__ void pack_w_kernel(const float* __restrict__ src, __bf16* __restrict__ dst, int KB){
  int idx = blockIdx.x * blockDim.x + threadIdx.x;
  int total = KB * 8 * 32 * 16;
  if (idx >= total) return;
  int i    = idx & 15;
  int lane = (idx >> 4) & 31;
  int jn   = (idx >> 9) & 7;
  int kb   = idx >> 12;
  int n    = jn*16 + (lane & 15);
  int half = lane >> 4;
  int j    = i >> 1;
  int k    = frag_k0(kb, half, j) + (i & 1);
  dst[idx] = (__bf16)src[(size_t)k * H_ + n];
}

// ---------------------------------------------------------------------------
// hh[t,n,h] = h[n,:16] @ embW + embB + time_emb[t]
// ---------------------------------------------------------------------------
__global__ void embed_kernel(const float* __restrict__ hin, const float* __restrict__ embW,
                             const float* __restrict__ embB, const float* __restrict__ temb,
                             float* __restrict__ hh){
  int idx = blockIdx.x * blockDim.x + threadIdx.x;
  int t = blockIdx.y;
  if (idx >= N_ * H_) return;
  int n = idx / H_, hc = idx % H_;
  float acc = embB[hc];
#pragma unroll
  for (int i = 0; i < IN_; ++i) acc += hin[n*IN_ + i] * embW[i*H_ + hc];
  hh[((size_t)t*N_ + n)*H_ + hc] = acc + temb[t*H_ + hc];
}

// ---------------------------------------------------------------------------
// GMN edge kernel: one wave per 16-edge tile (E/16 = 5000 tiles, exact).
//   m  = silu(silu(h[row]@We1a + h[col]@We1b + vec@We1v + rad*we1r + be1)@We2 + be2)
//   atomic: agg[row] += m ; xacc[row] += diff * (m.wx)
//   vec_out = vec_in + m@Wv
// ---------------------------------------------------------------------------
__global__ void gmn_edge_kernel(
    const float* __restrict__ hh, const float* __restrict__ x,
    const float* vec_in, float* vec_out,
    const int* __restrict__ row, const int* __restrict__ col,
    const unsigned short* __restrict__ We1a_p, const unsigned short* __restrict__ We1b_p,
    const unsigned short* __restrict__ We1v_p,
    const float* __restrict__ we1r, const float* __restrict__ be1,
    const unsigned short* __restrict__ We2_p, const float* __restrict__ be2,
    const unsigned short* __restrict__ Wv_p, const float* __restrict__ wx,
    float* __restrict__ agg, float* __restrict__ xacc)
{
  __shared__ __bf16 s_a[4][16*H_];
  __shared__ float  s_m[4][16*H_];
  __shared__ float  s_d[4][16*3];
  __shared__ float  s_r[4][16];

  const int t    = blockIdx.y;
  const int lane = threadIdx.x & 31;
  const int wv   = threadIdx.x >> 5;
  const int tile = blockIdx.x * 4 + wv;     // grid.x = E/64 -> always valid
  const int e0   = tile * 16;

  int ridx = 0, cidx = 0;
  if (lane < 16){ ridx = row[e0 + lane]; cidx = col[e0 + lane]; }

  // diff + radial (lanes 0..15, one edge each)
  if (lane < 16){
    float rad = 0.f;
#pragma unroll
    for (int d = 0; d < 3; ++d){
      float dd = x[((size_t)t*N_ + ridx)*3 + d] - x[((size_t)t*N_ + cidx)*3 + d];
      s_d[wv][lane*3 + d] = dd;
      rad += dd * dd;
    }
    s_r[wv][lane] = rad;
  }

  const int nc = lane & 15, half = lane >> 4;
  v8f acc[8];
#pragma unroll
  for (int jn = 0; jn < 8; ++jn)
#pragma unroll
    for (int r = 0; r < 8; ++r) acc[jn][r] = 0.0f;

  v16bf af[4];

  // ---- A = hh[row] ----
  for (int ii = 0; ii < 16; ++ii){
    int g = __shfl(ridx, ii, 32);
#pragma unroll
    for (int j = 0; j < 4; ++j){
      int hc = lane + 32*j;
      s_a[wv][ii*H_ + hc] = (__bf16)hh[((size_t)t*N_ + g)*H_ + hc];
    }
  }
  __syncthreads();
#pragma unroll
  for (int kb = 0; kb < 4; ++kb) af[kb] = afrag_lds_bf16(s_a[wv], lane, kb);
#pragma unroll
  for (int kb = 0; kb < 4; ++kb)
#pragma unroll
    for (int jn = 0; jn < 8; ++jn)
      acc[jn] = wmma_acc(af[kb], bfrag(We1a_p, lane, kb, jn), acc[jn]);
  __syncthreads();

  // ---- A = hh[col] ----
  for (int ii = 0; ii < 16; ++ii){
    int g = __shfl(cidx, ii, 32);
#pragma unroll
    for (int j = 0; j < 4; ++j){
      int hc = lane + 32*j;
      s_a[wv][ii*H_ + hc] = (__bf16)hh[((size_t)t*N_ + g)*H_ + hc];
    }
  }
  __syncthreads();
#pragma unroll
  for (int kb = 0; kb < 4; ++kb) af[kb] = afrag_lds_bf16(s_a[wv], lane, kb);
#pragma unroll
  for (int kb = 0; kb < 4; ++kb)
#pragma unroll
    for (int jn = 0; jn < 8; ++jn)
      acc[jn] = wmma_acc(af[kb], bfrag(We1b_p, lane, kb, jn), acc[jn]);
  __syncthreads();

  // ---- A = vec[t, e0..e0+15] ----
  for (int ii = 0; ii < 16; ++ii){
#pragma unroll
    for (int j = 0; j < 4; ++j){
      int hc = lane + 32*j;
      s_a[wv][ii*H_ + hc] = (__bf16)vec_in[((size_t)t*E_ + e0 + ii)*H_ + hc];
    }
  }
  __syncthreads();
#pragma unroll
  for (int kb = 0; kb < 4; ++kb) af[kb] = afrag_lds_bf16(s_a[wv], lane, kb);
#pragma unroll
  for (int kb = 0; kb < 4; ++kb)
#pragma unroll
    for (int jn = 0; jn < 8; ++jn)
      acc[jn] = wmma_acc(af[kb], bfrag(We1v_p, lane, kb, jn), acc[jn]);

  // ---- epilogue 1: + bias + radial rank-1, silu -> s_m ----
#pragma unroll
  for (int jn = 0; jn < 8; ++jn)
#pragma unroll
    for (int r = 0; r < 8; ++r){
      int mr = half*8 + r;
      int hc = jn*16 + nc;
      float v = acc[jn][r] + be1[hc] + s_r[wv][mr] * we1r[hc];
      s_m[wv][mr*H_ + hc] = silu_f(v);
    }
  __syncthreads();

  // ---- GEMM 2: m = silu(m @ We2 + be2) ----
#pragma unroll
  for (int kb = 0; kb < 4; ++kb) af[kb] = afrag_lds_f32(s_m[wv], lane, kb);
  v8f acc2[8];
#pragma unroll
  for (int jn = 0; jn < 8; ++jn)
#pragma unroll
    for (int r = 0; r < 8; ++r) acc2[jn][r] = 0.0f;
#pragma unroll
  for (int kb = 0; kb < 4; ++kb)
#pragma unroll
    for (int jn = 0; jn < 8; ++jn)
      acc2[jn] = wmma_acc(af[kb], bfrag(We2_p, lane, kb, jn), acc2[jn]);
  __syncthreads();
#pragma unroll
  for (int jn = 0; jn < 8; ++jn)
#pragma unroll
    for (int r = 0; r < 8; ++r){
      int mr = half*8 + r;
      int hc = jn*16 + nc;
      s_m[wv][mr*H_ + hc] = silu_f(acc2[jn][r] + be2[hc]);
    }
  __syncthreads();

  // ---- coordinate update: xacc[row] += diff * (m . wx) ----
  if (lane < 16){
    float c = 0.f;
    for (int hc = 0; hc < H_; ++hc) c += s_m[wv][lane*H_ + hc] * wx[hc];
#pragma unroll
    for (int d = 0; d < 3; ++d)
      atomicAdd(&xacc[((size_t)t*N_ + ridx)*3 + d], s_d[wv][lane*3 + d] * c);
  }

  // ---- GEMM 3: vec_out = vec_in + m @ Wv ; agg[row] += m ----
#pragma unroll
  for (int kb = 0; kb < 4; ++kb) af[kb] = afrag_lds_f32(s_m[wv], lane, kb);
  int garr[8];
#pragma unroll
  for (int r = 0; r < 8; ++r) garr[r] = __shfl(ridx, half*8 + r, 32);

#pragma unroll
  for (int jn = 0; jn < 8; ++jn){
    v8f a3;
#pragma unroll
    for (int r = 0; r < 8; ++r) a3[r] = 0.0f;
#pragma unroll
    for (int kb = 0; kb < 4; ++kb)
      a3 = wmma_acc(af[kb], bfrag(Wv_p, lane, kb, jn), a3);
#pragma unroll
    for (int r = 0; r < 8; ++r){
      int mr = half*8 + r;
      int hc = jn*16 + nc;
      float mval = s_m[wv][mr*H_ + hc];
      atomicAdd(&agg[((size_t)t*N_ + garr[r])*H_ + hc], mval);
      size_t vi = ((size_t)t*E_ + e0 + mr)*H_ + hc;
      vec_out[vi] = vec_in[vi] + a3[r];
    }
  }
}

// ---------------------------------------------------------------------------
// GMN node kernel: hn = silu(concat(h, agg)@Wh1 + bh1)@Wh2 + bh2 ; x += xacc
// One wave per 16-node tile; K=256 handled as two 128-K passes of packed Wh1.
// ---------------------------------------------------------------------------
__global__ void gmn_node_kernel(
    float* hh, const float* __restrict__ agg,
    float* x, const float* __restrict__ xacc,
    const unsigned short* __restrict__ Wh1_p, const float* __restrict__ bh1,
    const unsigned short* __restrict__ Wh2_p, const float* __restrict__ bh2)
{
  __shared__ __bf16 s_a[4][16*H_];
  __shared__ float  s_m[4][16*H_];

  const int t    = blockIdx.y;
  const int lane = threadIdx.x & 31;
  const int wv   = threadIdx.x >> 5;
  const int tile = blockIdx.x * 4 + wv;
  const int n0   = tile * 16;
  const bool tv  = (n0 < N_);

  const int nc = lane & 15, half = lane >> 4;
  v8f acc[8];
#pragma unroll
  for (int jn = 0; jn < 8; ++jn)
#pragma unroll
    for (int r = 0; r < 8; ++r) acc[jn][r] = 0.0f;
  v16bf af[4];

  // ---- stage h rows, K 0..127 of Wh1 ----
  for (int ii = 0; ii < 16; ++ii){
    int n = n0 + ii;
    bool vrow = tv && (n < N_);
#pragma unroll
    for (int j = 0; j < 4; ++j){
      int hc = lane + 32*j;
      s_a[wv][ii*H_ + hc] = vrow ? (__bf16)hh[((size_t)t*N_ + n)*H_ + hc] : (__bf16)0.0f;
    }
  }
  __syncthreads();
#pragma unroll
  for (int kb = 0; kb < 4; ++kb) af[kb] = afrag_lds_bf16(s_a[wv], lane, kb);
#pragma unroll
  for (int kb = 0; kb < 4; ++kb)
#pragma unroll
    for (int jn = 0; jn < 8; ++jn)
      acc[jn] = wmma_acc(af[kb], bfrag(Wh1_p, lane, kb, jn), acc[jn]);
  __syncthreads();

  // ---- stage agg rows, K 128..255 of Wh1 ----
  for (int ii = 0; ii < 16; ++ii){
    int n = n0 + ii;
    bool vrow = tv && (n < N_);
#pragma unroll
    for (int j = 0; j < 4; ++j){
      int hc = lane + 32*j;
      s_a[wv][ii*H_ + hc] = vrow ? (__bf16)agg[((size_t)t*N_ + n)*H_ + hc] : (__bf16)0.0f;
    }
  }
  __syncthreads();
#pragma unroll
  for (int kb = 0; kb < 4; ++kb) af[kb] = afrag_lds_bf16(s_a[wv], lane, kb);
#pragma unroll
  for (int kb = 0; kb < 4; ++kb)
#pragma unroll
    for (int jn = 0; jn < 8; ++jn)
      acc[jn] = wmma_acc(af[kb], bfrag(Wh1_p, lane, 4 + kb, jn), acc[jn]);

  // ---- epilogue: silu -> s_m ----
#pragma unroll
  for (int jn = 0; jn < 8; ++jn)
#pragma unroll
    for (int r = 0; r < 8; ++r){
      int mr = half*8 + r;
      int hc = jn*16 + nc;
      s_m[wv][mr*H_ + hc] = silu_f(acc[jn][r] + bh1[hc]);
    }
  __syncthreads();

  // ---- Wh2 GEMM + bias, store ----
#pragma unroll
  for (int kb = 0; kb < 4; ++kb) af[kb] = afrag_lds_f32(s_m[wv], lane, kb);
#pragma unroll
  for (int jn = 0; jn < 8; ++jn){
    v8f a2;
#pragma unroll
    for (int r = 0; r < 8; ++r) a2[r] = 0.0f;
#pragma unroll
    for (int kb = 0; kb < 4; ++kb)
      a2 = wmma_acc(af[kb], bfrag(Wh2_p, lane, kb, jn), a2);
#pragma unroll
    for (int r = 0; r < 8; ++r){
      int mr = half*8 + r;
      int n = n0 + mr;
      if (tv && n < N_)
        hh[((size_t)t*N_ + n)*H_ + jn*16 + nc] = a2[r] + bh2[jn*16 + nc];
    }
  }

  // ---- x update ----
  if (lane < 16){
    int n = n0 + lane;
    if (tv && n < N_){
#pragma unroll
      for (int d = 0; d < 3; ++d)
        x[((size_t)t*N_ + n)*3 + d] += xacc[((size_t)t*N_ + n)*3 + d];
    }
  }
}

// ---------------------------------------------------------------------------
// Generic row-major GEMM: out[M,128] = in[M,128] @ Wpacked  (A frags direct
// from global; one wave per 16-row tile; used for q/k/v projections)
// ---------------------------------------------------------------------------
__global__ void gemm_rm_kernel(const float* __restrict__ in,
                               const unsigned short* __restrict__ wp,
                               float* __restrict__ out, int M){
  const int lane = threadIdx.x & 31;
  const int wv   = threadIdx.x >> 5;
  const int tile = blockIdx.x * 4 + wv;
  const int m0   = tile * 16;
  if (m0 >= M) return;
  const int mrow = lane & 15, half = lane >> 4;
  const float* arow = in + (size_t)(m0 + mrow) * H_;
  v16bf af[4];
#pragma unroll
  for (int kb = 0; kb < 4; ++kb){
    v16bf a;
#pragma unroll
    for (int j = 0; j < 8; ++j){
      int k0 = frag_k0(kb, half, j);
      a[2*j]   = (__bf16)arow[k0];
      a[2*j+1] = (__bf16)arow[k0+1];
    }
    af[kb] = a;
  }
  const int nc = lane & 15;
#pragma unroll
  for (int jn = 0; jn < 8; ++jn){
    v8f acc;
#pragma unroll
    for (int r = 0; r < 8; ++r) acc[r] = 0.0f;
#pragma unroll
    for (int kb = 0; kb < 4; ++kb)
      acc = wmma_acc(af[kb], bfrag(wp, lane, kb, jn), acc);
#pragma unroll
    for (int r = 0; r < 8; ++r){
      int mr = half*8 + r;
      out[(size_t)(m0 + mr)*H_ + jn*16 + nc] = acc[r];
    }
  }
}

// ---------------------------------------------------------------------------
// Temporal attention (T=10) per node: scores/softmax/mix. q,k,v precomputed.
// ---------------------------------------------------------------------------
__global__ void attn_kernel(const float* __restrict__ q, const float* __restrict__ k,
                            const float* __restrict__ v, float* hh, float* x){
  const int n = blockIdx.x;
  const int tid = threadIdx.x;
  __shared__ float sq[T_][H_], sk[T_][H_], sv[T_][H_];
  __shared__ float sc[T_][T_];
  __shared__ float sx[T_][3], sxl[3];

  for (int t = 0; t < T_; ++t){
    size_t o = ((size_t)t*N_ + n)*H_ + tid;
    sq[t][tid] = q[o]; sk[t][tid] = k[o]; sv[t][tid] = v[o];
  }
  if (tid < T_*3){ int t = tid/3, d = tid%3; sx[t][d] = x[((size_t)t*N_ + n)*3 + d]; }
  if (tid < 3)    sxl[tid] = x[((size_t)(T_-1)*N_ + n)*3 + tid];
  __syncthreads();

  if (tid < T_*T_){
    int tt = tid / T_, ss = tid % T_;
    float d = 0.f;
    for (int hc = 0; hc < H_; ++hc) d += sq[tt][hc] * sk[ss][hc];
    sc[tt][ss] = d * (1.0f / sqrtf((float)H_));
  }
  __syncthreads();
  if (tid < T_){
    float mx = -1e30f;
    for (int ss = 0; ss < T_; ++ss) mx = fmaxf(mx, sc[tid][ss]);
    float sum = 0.f;
    for (int ss = 0; ss < T_; ++ss){ float e = __expf(sc[tid][ss] - mx); sc[tid][ss] = e; sum += e; }
    float inv = 1.0f / sum;
    for (int ss = 0; ss < T_; ++ss) sc[tid][ss] *= inv;
  }
  __syncthreads();

  for (int tt = 0; tt < T_; ++tt){
    float o = 0.f;
    for (int ss = 0; ss < T_; ++ss) o += sc[tt][ss] * sv[ss][tid];
    hh[((size_t)tt*N_ + n)*H_ + tid] = o;
  }
  if (tid < T_*3){
    int tt = tid/3, d = tid%3;
    float o = sxl[d];
    for (int ss = 0; ss < T_; ++ss) o += sc[tt][ss] * (sx[ss][d] - sxl[d]);
    x[((size_t)tt*N_ + n)*3 + d] = o;
  }
}

// ---------------------------------------------------------------------------
// x_hat[f,n,d] = x_last + sum_t theta[f,t]*(x[t]-x_last)
// ---------------------------------------------------------------------------
__global__ void pool_kernel(const float* __restrict__ x, const float* __restrict__ theta,
                            float* __restrict__ out){
  int idx = blockIdx.x*blockDim.x + threadIdx.x;
  if (idx >= F_*N_*3) return;
  int f = idx / (N_*3);
  int rem = idx % (N_*3);
  int n = rem / 3, d = rem % 3;
  float xl = x[((size_t)(T_-1)*N_ + n)*3 + d];
  float o = xl;
  for (int t = 0; t < T_; ++t)
    o += theta[f*T_ + t] * (x[((size_t)t*N_ + n)*3 + d] - xl);
  out[idx] = o;
}

// ---------------------------------------------------------------------------
// Host launcher
// ---------------------------------------------------------------------------
struct GmnW { const float *We1,*be1,*We2,*be2,*wx,*Wv,*Wh1,*bh1,*Wh2,*bh2; };
struct GmnP { unsigned short *a,*b,*v,*e2,*wv,*h1,*h2; };

extern "C" void kernel_launch(void* const* d_in, const int* in_sizes, int n_in,
                              void* d_out, int out_size, void* d_ws, size_t ws_size,
                              hipStream_t stream) {
  (void)in_sizes; (void)n_in; (void)out_size; (void)ws_size;

  // Input layout (setup_inputs dict insertion order, params flattened recursively):
  // 0:h 1:x 2:edges 3:edge_attr(unused) 4:vec 5:cfg(unused)
  // 6:emb_W 7:emb_b 8:time_emb
  // 9-18:gmn1{We1,be1,We2,be2,wx,Wv,Wh1,bh1,Wh2,bh2} 19-28:gmn2 29-38:gmnt1_gmn
  // 39-41:gmnt1_att{Wq,Wk,Wo} 42-51:gmnt2_gmn 52-54:gmnt2_att 55:theta
  auto F32 = [&](int i){ return (const float*)d_in[i]; };
  const float* h_in   = F32(0);
  const float* x_in   = F32(1);
  const int*   edges  = (const int*)d_in[2];
  const float* vec_in0= F32(4);
  const float* embW   = F32(6);
  const float* embB   = F32(7);
  const float* temb   = F32(8);
  const float* theta  = F32(55);

  auto mkg = [&](int b){
    GmnW g; g.We1=F32(b); g.be1=F32(b+1); g.We2=F32(b+2); g.be2=F32(b+3);
    g.wx=F32(b+4); g.Wv=F32(b+5); g.Wh1=F32(b+6); g.bh1=F32(b+7);
    g.Wh2=F32(b+8); g.bh2=F32(b+9); return g; };
  GmnW g1 = mkg(9), g2 = mkg(19), gt1 = mkg(29), gt2 = mkg(42);

  // ---- workspace carve ----
  size_t o = 0;
  auto carve = [&](size_t bytes){ size_t r = o; o = (o + bytes + 255) & ~(size_t)255; return r; };
  const size_t TNH = (size_t)T_*N_*H_*sizeof(float);   // 25.6 MB
  const size_t TEH = (size_t)T_*E_*H_*sizeof(float);   // 409.6 MB
  const size_t TN3 = (size_t)T_*N_*3*sizeof(float);
  char* ws = (char*)d_ws;
  float* hhp   = (float*)(ws + carve(TNH));
  float* vecws = (float*)(ws + carve(TEH));
  float* xws   = (float*)(ws + carve(TN3));
  float* aggq  = (float*)(ws + carve(TNH));   // agg during GMN, q during attn
  float* kbuf  = (float*)(ws + carve(TNH));
  float* vbuf  = (float*)(ws + carve(TNH));
  float* xacc  = (float*)(ws + carve(TN3));
  unsigned short* wpack = (unsigned short*)(ws + carve((size_t)2*1024*1024));

  // ---- weight packing (every call; deterministic) ----
  size_t po = 0;
  auto packmat = [&](const float* src, int KB)->unsigned short*{
    unsigned short* dst = wpack + po;
    int total = KB*8*32*16;
    po += (size_t)total;
    pack_w_kernel<<<dim3((total+255)/256), dim3(256), 0, stream>>>(src, (__bf16*)dst, KB);
    return dst;
  };
  auto packgmn = [&](const GmnW& g)->GmnP{
    GmnP p;
    p.a  = packmat(g.We1,             4);   // feat cols   0..127 (h[row])
    p.b  = packmat(g.We1 + 128*H_,    4);   // feat cols 128..255 (h[col])
    p.v  = packmat(g.We1 + 257*H_,    4);   // feat cols 257..384 (vec)
    p.e2 = packmat(g.We2,             4);
    p.wv = packmat(g.Wv,              4);
    p.h1 = packmat(g.Wh1,             8);   // K = 256
    p.h2 = packmat(g.Wh2,             4);
    return p;
  };
  GmnP p1 = packgmn(g1), p2 = packgmn(g2), pt1 = packgmn(gt1), pt2 = packgmn(gt2);

  // ---- init activations ----
  hipMemcpyAsync(xws, x_in, TN3, hipMemcpyDeviceToDevice, stream);
  embed_kernel<<<dim3((N_*H_+255)/256, T_), dim3(256), 0, stream>>>(h_in, embW, embB, temb, hhp);

  // ---- GMN layer runner ----
  auto run_gmn = [&](const GmnW& g, const GmnP& p, const float* vin, float* vout){
    hipMemsetAsync(aggq, 0, TNH, stream);
    hipMemsetAsync(xacc, 0, TN3, stream);
    gmn_edge_kernel<<<dim3(E_/64, T_), dim3(128), 0, stream>>>(
        hhp, xws, vin, vout, edges, edges + E_,
        p.a, p.b, p.v, g.We1 + 256*H_, g.be1, p.e2, g.be2, p.wv, g.wx, aggq, xacc);
    const int NT = (N_ + 15)/16;  // 313
    gmn_node_kernel<<<dim3((NT+3)/4, T_), dim3(128), 0, stream>>>(
        hhp, aggq, xws, xacc, p.h1, g.bh1, p.h2, g.bh2);
  };

  // ---- attention runner ----
  auto run_att = [&](int b){
    unsigned short* qp = packmat(F32(b),   4);
    unsigned short* kp = packmat(F32(b+1), 4);
    unsigned short* op = packmat(F32(b+2), 4);
    const int M = T_*N_;
    const int blocks = ((M + 15)/16 + 3)/4;
    gemm_rm_kernel<<<dim3(blocks), dim3(128), 0, stream>>>(hhp, qp, aggq, M);
    gemm_rm_kernel<<<dim3(blocks), dim3(128), 0, stream>>>(hhp, kp, kbuf, M);
    gemm_rm_kernel<<<dim3(blocks), dim3(128), 0, stream>>>(hhp, op, vbuf, M);
    attn_kernel<<<dim3(N_), dim3(128), 0, stream>>>(aggq, kbuf, vbuf, hhp, xws);
  };

  // ---- pipeline (reference order) ----
  run_gmn(g1,  p1,  vec_in0, vecws);
  run_gmn(gt1, pt1, vecws,   vecws);
  run_att(39);
  run_gmn(g2,  p2,  vecws,   vecws);
  run_gmn(gt2, pt2, vecws,   vecws);
  run_att(52);

  pool_kernel<<<dim3((F_*N_*3+255)/256), dim3(256), 0, stream>>>(xws, theta, (float*)d_out);
}
